// AdaptAttention_79190607004167
// MI455X (gfx1250) — matrix operand order repaired and checked by reading, not compiled, not run on hardware
//
#include <hip/hip_runtime.h>
#include <hip/hip_bf16.h>
#include <math.h>

typedef _Float16 half_t;
typedef __attribute__((ext_vector_type(16))) _Float16 v16h;
typedef __attribute__((ext_vector_type(8)))  _Float16 v8h;
typedef __attribute__((ext_vector_type(8)))  float    v8f;
typedef __attribute__((ext_vector_type(4)))  float    v4f;
typedef __attribute__((ext_vector_type(4)))  unsigned v4u;

#define BN   1024            // sequence length N
#define DKK  64              // head dim
#define NH   8               // total heads
#define NB   16              // batch
#define HLOC 4               // local heads
#define QKV_ELEMS (NB*NH*BN*DKK)   // 8,388,608
#define SSTR 1028            // padded LDS score row stride (16B aligned, 4-bank skew)

// ---------------------------------------------------------------------------
// Prep 1: fp32 -> f16 with LDS tile-transpose so ALL global traffic is
// contiguous 32B per lane.  Qh = Q/sqrt(dk);  Kh = K;  Vt = V^T (dk-major).
// One block = one 64x64 (n x d) tile of one (b,h).
// ---------------------------------------------------------------------------
__global__ __launch_bounds__(256)
void k_convert(const float* __restrict__ Q, const float* __restrict__ K,
               const float* __restrict__ V,
               half_t* __restrict__ Qh, half_t* __restrict__ Kh,
               half_t* __restrict__ Vt)
{
    __shared__ float tv[64][65];                     // padded: conflict-free cols
    const int bh = blockIdx.x >> 4;
    const int nt = blockIdx.x & 15;
    const int t  = threadIdx.x;
    const size_t base = ((size_t)bh * BN + nt * 64) * DKK;   // 4096 elems

    {   // each thread: 16 consecutive elems (one quarter-row)
        const int e0 = t * 16;
        const int row = e0 >> 6, dc = e0 & 63;
        const float* qp = Q + base + e0;
        const float* kp = K + base + e0;
        const float* vp = V + base + e0;
        v16h qh, kh;
        #pragma unroll
        for (int j = 0; j < 16; ++j) {
            qh[j] = (half_t)(qp[j] * 0.125f);        // fold 1/sqrt(64)
            kh[j] = (half_t)kp[j];
            tv[row][dc + j] = vp[j];
        }
        *(v16h*)(Qh + base + e0) = qh;
        *(v16h*)(Kh + base + e0) = kh;
    }
    __syncthreads();
    {   // transposed write: lane owns 16 consecutive n for one d
        const int d = t >> 2, part = t & 3;
        v16h hv;
        #pragma unroll
        for (int j = 0; j < 16; ++j)
            hv[j] = (half_t)tv[part * 16 + j][d];
        *(v16h*)(Vt + ((size_t)bh * DKK + d) * BN + nt * 64 + part * 16) = hv;
    }
}

// ---------------------------------------------------------------------------
// Prep 2: pack mask to 1 bit/element with wave32 ballot (2 MB total).
// ---------------------------------------------------------------------------
__global__ __launch_bounds__(256)
void k_maskpack(const int* __restrict__ mask, unsigned* __restrict__ mbits)
{
    int t = blockIdx.x * 256 + threadIdx.x;
    unsigned bits = (unsigned)__ballot(mask[t] != 0);
    if ((threadIdx.x & 31) == 0) mbits[t >> 5] = bits;
}

// ---------------------------------------------------------------------------
// Prep 3: fold mlp_w into user_proj_w (w2) and rel_pos_table (relw[hl][idx],
// transposed for unit-stride coalesced reads in the score loop).
// ---------------------------------------------------------------------------
__global__ __launch_bounds__(256)
void k_w2rel(const float* __restrict__ upw, const float* __restrict__ relt,
             const float* __restrict__ mlpw,
             float* __restrict__ w2, float* __restrict__ relw)
{
    int t = blockIdx.x * 256 + threadIdx.x;
    if (t < 2048) {
        int c = t >> 2, hl = t & 3;
        float s = 0.f;
        #pragma unroll
        for (int d = 0; d < DKK; ++d)
            s += upw[c * 256 + hl * 64 + d] * mlpw[hl * 64 + d];
        w2[t] = s;
    } else {
        int u = t - 2048;
        int i = u >> 2, hl = u & 3;
        if (i < 2 * BN - 1) {
            float s = 0.f;
            #pragma unroll
            for (int d = 0; d < DKK; ++d)
                s += relt[(size_t)i * 256 + hl * 64 + d] * mlpw[hl * 64 + d];
            relw[hl * 2048 + i] = s;
        }
    }
}

// ---------------------------------------------------------------------------
// Prep 4: per-position reweight scalars rw_v, rw_u.
// ---------------------------------------------------------------------------
__global__ __launch_bounds__(256)
void k_rw(const float* __restrict__ V, const float* __restrict__ users,
          const float* __restrict__ upb, const float* __restrict__ mlpw,
          const float* __restrict__ w2,
          float* __restrict__ rw_v, float* __restrict__ rw_u)
{
    int t = blockIdx.x * 256 + threadIdx.x;          // 65536
    int b  = t >> 12;
    int hl = (t >> 10) & 3;
    int n  = t & 1023;

    const float* mw = mlpw + hl * DKK;
    const float* vp = V + (((size_t)(b * NH + (NH - HLOC) + hl)) * BN + n) * DKK;
    float sv = 0.f;
    #pragma unroll
    for (int d = 0; d < DKK; ++d) sv += vp[d] * mw[d];
    rw_v[t] = sv;

    const float* up = users + ((size_t)b * BN + n) * 512;
    float su = 0.f;
    for (int c = 0; c < 512; ++c) su += up[c] * w2[c * 4 + hl];
    float sb = 0.f;
    #pragma unroll
    for (int d = 0; d < DKK; ++d) sb += upb[hl * DKK + d] * mw[d];
    rw_u[t] = su + sb;
}

// ---------------------------------------------------------------------------
// Main fused attention.  128 thr = 4 waves; each wave owns a 16-query strip
// of a 64-query tile of one (b,h).  Scores in LDS (64x1028 fp32).  Both WMMA
// loops are software-pipelined: next iteration's global fragments are in
// flight while the current WMMAs execute (1 wave/SIMD at this LDS size).
// ---------------------------------------------------------------------------
__global__ __launch_bounds__(128)
void k_attn(const half_t* __restrict__ Qh, const half_t* __restrict__ Kh,
            const half_t* __restrict__ Vt, const unsigned* __restrict__ mbits,
            const float* __restrict__ rw_v, const float* __restrict__ rw_u,
            const float* __restrict__ relw, const float* __restrict__ mlp_b,
            float* __restrict__ out, float* __restrict__ pg,
            float* __restrict__ pl)
{
    extern __shared__ float sS[];                    // [64][SSTR]
    const int tid  = threadIdx.x;
    const int wave = tid >> 5;
    const int lane = tid & 31;
    const int ln   = lane & 15;
    const int hi   = lane >> 4;
    const int h    = blockIdx.y;
    const int b    = blockIdx.z;
    const int q0   = blockIdx.x * 64;
    const int r0   = wave * 16;
    const int bh   = b * NH + h;

    const bool isLoc = (h >= NH - HLOC);
    const int  hl    = isLoc ? (h - (NH - HLOC)) : 0;

    // ---- Q A-fragments (contiguous k = 16*hi + s, same as K B-frags), both dk halves ----
    const half_t* qrow = Qh + ((size_t)bh * BN + q0 + r0 + ln) * DKK;
    v16h aq0, aq1;
    {
        v8h l0 = *(const v8h*)(qrow +      hi * 16);
        v8h h0 = *(const v8h*)(qrow +  8 + hi * 16);
        v8h l1 = *(const v8h*)(qrow + 32 + hi * 16);
        v8h h1 = *(const v8h*)(qrow + 40 + hi * 16);
        #pragma unroll
        for (int j = 0; j < 8; ++j) {
            aq0[j] = l0[j]; aq0[8 + j] = h0[j];
            aq1[j] = l1[j]; aq1[8 + j] = h1[j];
        }
    }

    // ---- S = Qh * K^T (+reweight) -> LDS; pipelined, two uniform variants --
    const half_t* kcol = Kh + (size_t)bh * BN * DKK + (size_t)ln * DKK + hi * 16;
    v16h kc0 = *(const v16h*)(kcol);
    v16h kc1 = *(const v16h*)(kcol + 32);
    if (isLoc) {
        const float* rwv = rw_v + (size_t)(b * HLOC + hl) * BN;
        const float* rwu = rw_u + (size_t)(b * HLOC + hl) * BN;
        const float  mb  = mlp_b[hl];
        const int qbase = q0 + r0 + 8 * hi;
        float rowc[8];
        #pragma unroll
        for (int v = 0; v < 8; ++v)
            rowc[v] = rwv[qbase + v] + rwu[qbase + v] + mb;
        const float* relq = relw + hl * 2048 + (BN - 1 - qbase);
        #pragma unroll 2
        for (int ct = 0; ct < 64; ++ct) {
            v16h nx0 = kc0, nx1 = kc1;
            if (ct < 63) {                           // preload next fragments
                nx0 = *(const v16h*)(kcol + (size_t)(ct + 1) * 16 * DKK);
                nx1 = *(const v16h*)(kcol + (size_t)(ct + 1) * 16 * DKK + 32);
            }
            v8f c = {};
            c = __builtin_amdgcn_wmma_f32_16x16x32_f16(false, aq0, false, kc0,
                                                       (short)0, c, false, false);
            c = __builtin_amdgcn_wmma_f32_16x16x32_f16(false, aq1, false, kc1,
                                                       (short)0, c, false, false);
            const int col = ct * 16 + ln;
            const float cterm = rwv[col];
            const float* relc = relq + col;
            #pragma unroll
            for (int v = 0; v < 8; ++v)              // C: M=v+8*hi, N=ln
                sS[(size_t)(r0 + 8 * hi + v) * SSTR + col] =
                    c[v] + rowc[v] + relc[-v] + cterm;
            kc0 = nx0; kc1 = nx1;
        }
    } else {
        #pragma unroll 2
        for (int ct = 0; ct < 64; ++ct) {
            v16h nx0 = kc0, nx1 = kc1;
            if (ct < 63) {
                nx0 = *(const v16h*)(kcol + (size_t)(ct + 1) * 16 * DKK);
                nx1 = *(const v16h*)(kcol + (size_t)(ct + 1) * 16 * DKK + 32);
            }
            v8f c = {};
            c = __builtin_amdgcn_wmma_f32_16x16x32_f16(false, aq0, false, kc0,
                                                       (short)0, c, false, false);
            c = __builtin_amdgcn_wmma_f32_16x16x32_f16(false, aq1, false, kc1,
                                                       (short)0, c, false, false);
            const int col = ct * 16 + ln;
            #pragma unroll
            for (int v = 0; v < 8; ++v)
                sS[(size_t)(r0 + 8 * hi + v) * SSTR + col] = c[v];
            kc0 = nx0; kc1 = nx1;
        }
    }

    // ---- masked softmax per row; 16B-vectorized LDS + global traffic ----
    {
        const int row = r0 + ln;
        const int q   = q0 + row;
        v4f* s4 = (v4f*)(sS + (size_t)row * SSTR + hi * 512);
        const v4u* m4 = (const v4u*)(mbits + ((size_t)b * BN + q) * 32 + hi * 16);
        v4u mv[4];
        #pragma unroll
        for (int g = 0; g < 4; ++g) mv[g] = m4[g];   // 4 x b128 mask loads

        float mx = -INFINITY;
        #pragma unroll
        for (int g = 0; g < 4; ++g) {
            #pragma unroll
            for (int cc = 0; cc < 4; ++cc) {
                const unsigned bits = mv[g][cc];
                const int wbase = (g * 4 + cc) * 8;
                for (int u = 0; u < 8; ++u) {
                    v4f x = s4[wbase + u];
                    x.x = (bits >> (u * 4 + 0)) & 1u ? x.x : -1.0e12f;
                    x.y = (bits >> (u * 4 + 1)) & 1u ? x.y : -1.0e12f;
                    x.z = (bits >> (u * 4 + 2)) & 1u ? x.z : -1.0e12f;
                    x.w = (bits >> (u * 4 + 3)) & 1u ? x.w : -1.0e12f;
                    s4[wbase + u] = x;               // keep masked value
                    mx = fmaxf(mx, fmaxf(fmaxf(x.x, x.y), fmaxf(x.z, x.w)));
                }
            }
        }
        mx = fmaxf(mx, __shfl_xor(mx, 16));

        float sum = 0.f;
        for (int u = 0; u < 128; ++u) {
            v4f x = s4[u];
            x.x = __expf(x.x - mx); x.y = __expf(x.y - mx);
            x.z = __expf(x.z - mx); x.w = __expf(x.w - mx);
            s4[u] = x;
            sum += (x.x + x.y) + (x.z + x.w);
        }
        sum += __shfl_xor(sum, 16);
        const float inv = 1.f / sum;

        float* pout = isLoc
            ? pl + (((size_t)(b * HLOC + hl)) * BN + q) * BN
            : pg + (((size_t)(b * HLOC + h )) * BN + q) * BN;
        v4f* p4 = (v4f*)(pout + hi * 512);
        for (int u = 0; u < 128; ++u) {
            v4f x = s4[u] * inv;
            s4[u] = x;                               // normalized P for PV
            __builtin_nontemporal_store(x, &p4[u]);  // write-once: TH=NT
        }
    }

    // ---- O = P * V  (4 independent WMMA chains; pipelined V loads) ----
    v8f o0 = {}, o1 = {}, o2 = {}, o3 = {};
    const half_t* vtb = Vt + (size_t)bh * DKK * BN + hi * 16;
    const float*  prow = sS + (size_t)(r0 + ln) * SSTR;
    v16h vb0 = *(const v16h*)(vtb + (size_t)(     ln) * BN);
    v16h vb1 = *(const v16h*)(vtb + (size_t)(16 + ln) * BN);
    v16h vb2 = *(const v16h*)(vtb + (size_t)(32 + ln) * BN);
    v16h vb3 = *(const v16h*)(vtb + (size_t)(48 + ln) * BN);
    for (int kb = 0; kb < BN; kb += 32) {
        v16h n0 = vb0, n1 = vb1, n2 = vb2, n3 = vb3;
        if (kb < BN - 32) {                          // preload next V fragments
            const size_t koff = (size_t)kb + 32;
            n0 = *(const v16h*)(vtb + (size_t)(     ln) * BN + koff);
            n1 = *(const v16h*)(vtb + (size_t)(16 + ln) * BN + koff);
            n2 = *(const v16h*)(vtb + (size_t)(32 + ln) * BN + koff);
            n3 = *(const v16h*)(vtb + (size_t)(48 + ln) * BN + koff);
        }
        const v4f* pk4 = (const v4f*)(prow + kb + hi * 16);  // 16B aligned
        v4f a0 = pk4[0], a1 = pk4[1];                // K = 16*hi + 0..7
        v4f a2 = pk4[2], a3 = pk4[3];                // K = 16*hi + 8..15
        v16h ap;
        #pragma unroll
        for (int j = 0; j < 4; ++j) {
            ap[j]      = (half_t)a0[j];  ap[4 + j]  = (half_t)a1[j];
            ap[8 + j]  = (half_t)a2[j];  ap[12 + j] = (half_t)a3[j];
        }
        o0 = __builtin_amdgcn_wmma_f32_16x16x32_f16(false, ap, false, vb0,
                                                    (short)0, o0, false, false);
        o1 = __builtin_amdgcn_wmma_f32_16x16x32_f16(false, ap, false, vb1,
                                                    (short)0, o1, false, false);
        o2 = __builtin_amdgcn_wmma_f32_16x16x32_f16(false, ap, false, vb2,
                                                    (short)0, o2, false, false);
        o3 = __builtin_amdgcn_wmma_f32_16x16x32_f16(false, ap, false, vb3,
                                                    (short)0, o3, false, false);
        vb0 = n0; vb1 = n1; vb2 = n2; vb3 = n3;
    }
    float* orow = out + ((size_t)bh * BN + q0 + r0) * DKK;
    #pragma unroll
    for (int v = 0; v < 8; ++v) {
        float* op = orow + (size_t)(v + 8 * hi) * DKK;
        __builtin_nontemporal_store(o0[v], op +      ln);
        __builtin_nontemporal_store(o1[v], op + 16 + ln);
        __builtin_nontemporal_store(o2[v], op + 32 + ln);
        __builtin_nontemporal_store(o3[v], op + 48 + ln);
    }
}

// ---------------------------------------------------------------------------
extern "C" void kernel_launch(void* const* d_in, const int* in_sizes, int n_in,
                              void* d_out, int out_size, void* d_ws, size_t ws_size,
                              hipStream_t stream)
{
    const float* Q     = (const float*)d_in[0];
    const float* K     = (const float*)d_in[1];
    const float* V     = (const float*)d_in[2];
    const int*   M     = (const int*)  d_in[3];
    const float* users = (const float*)d_in[4];
    const float* relt  = (const float*)d_in[5];
    const float* upw   = (const float*)d_in[6];
    const float* upb   = (const float*)d_in[7];
    const float* mlpw  = (const float*)d_in[8];
    const float* mlpb  = (const float*)d_in[9];

    float* out = (float*)d_out;                       // [16,8,1024,64]
    float* pg  = out + (size_t)NB * NH * BN * DKK;    // [16,4,1024,1024]
    float* pl  = pg  + (size_t)NB * HLOC * BN * BN;   // [16,4,1024,1024]

    // workspace layout
    half_t*   Qh  = (half_t*)d_ws;
    half_t*   Kh  = Qh + QKV_ELEMS;
    half_t*   Vt  = Kh + QKV_ELEMS;
    float*    rwv = (float*)(Vt + QKV_ELEMS);
    float*    rwu = rwv + NB * HLOC * BN;
    float*    rlw = rwu + NB * HLOC * BN;             // 4 x 2048
    float*    w2  = rlw + 4 * 2048;
    unsigned* mb  = (unsigned*)(w2 + 2048);           // packed mask, 2 MB

    k_convert<<<NB * NH * (BN / 64), 256, 0, stream>>>(Q, K, V, Qh, Kh, Vt);
    k_maskpack<<<(NB * BN * BN) / 256, 256, 0, stream>>>(M, mb);
    k_w2rel<<<(2048 + 4 * (2 * BN - 1) + 255) / 256, 256, 0, stream>>>(
        upw, relt, mlpw, w2, rlw);
    k_rw<<<(NB * HLOC * BN) / 256, 256, 0, stream>>>(V, users, upb, mlpw, w2,
                                                     rwv, rwu);

    const int shmem = 64 * SSTR * sizeof(float);      // ~257 KB (<320 KB WGP LDS)
    hipFuncSetAttribute((const void*)k_attn,
                        hipFuncAttributeMaxDynamicSharedMemorySize, shmem);
    dim3 grid(BN / 64, NH, NB);                       // (16, 8, 16)
    k_attn<<<grid, 128, shmem, stream>>>(Qh, Kh, Vt, mb, rwv, rwu, rlw, mlpb,
                                         out, pg, pl);
}
